// GraphConvMx_29420525977638
// MI455X (gfx1250) — matrix operand order repaired and checked by reading, not compiled, not run on hardware
//
#include <hip/hip_runtime.h>
#include <hip/hip_bf16.h>
#include <stdint.h>

// Problem constants (fixed by the reference)
#define Bsz 8
#define Nn  6144
#define Dd  16
#define Kk  3
#define Oo  64
#define Cc  128          // B*D columns of the big GEMM
#define MBk 192          // Nn / 32 K-chunks
#define ROWS_PER_WG 64
#define WG_THREADS  128  // 4 waves (wave32)

typedef __attribute__((ext_vector_type(16))) __bf16 v16bf;
typedef __attribute__((ext_vector_type(8)))  float  v8f;
typedef __attribute__((ext_vector_type(4)))  float  v4f;   // native clang vector (nontemporal-OK)

union BF16x16 {
    v16bf v;
    unsigned short u[16];
    uint4 q[2];
};

__device__ __forceinline__ unsigned short f2bf(float f) {
    // round-to-nearest-even float -> bf16
    unsigned int u = __float_as_uint(f);
    u += 0x7FFFu + ((u >> 16) & 1u);
    return (unsigned short)(u >> 16);
}

__device__ __forceinline__ v4f ldnt4(const float* p) {
    // sup is streamed exactly once: non-temporal so it doesn't evict Xfrag/Y in L2
    return __builtin_nontemporal_load((const v4f*)p);
}

// ---------------------------------------------------------------------------
// Kernel 1: pack X (inputs [B,N,D] f32) into WMMA B-fragment order, bf16.
// Fragment index: frag = (mb*8 + g)*32 + lane ; 16 bf16 (32B) per lane.
// B-matrix 32x16 layout: lane = col (0..15) | half*16 ; lane half selects
// K 0..7,16..23 vs K 8..15,24..31 (j<8 / j>=8). col c = 16*g + (lane&15) -> b = g, d = lane&15.
// ---------------------------------------------------------------------------
__global__ __launch_bounds__(256)
void pack_bfrag_kernel(const float* __restrict__ inputs,
                       unsigned short* __restrict__ xfrag) {
    int t = blockIdx.x * 256 + threadIdx.x;       // 192*8*32 = 49152 threads
    int lane = t & 31;
    int g    = (t >> 5) & 7;                      // col group == batch b
    int mb   = t >> 8;                            // K-chunk
    if (mb >= MBk) return;
    int d  = lane & 15;
    int m0 = mb * 32 + (lane >> 4) * 8;           // lane half -> K 0..7,16..23 / 8..15,24..31
    const float* src = inputs + ((size_t)g * Nn + m0) * Dd + d;
    BF16x16 f;
#pragma unroll
    for (int j = 0; j < 16; ++j)
        f.u[j] = f2bf(src[(size_t)(j < 8 ? j : j + 8) * Dd]);
    uint4* dst = (uint4*)(xfrag + (size_t)t * 16);
    dst[0] = f.q[0];
    dst[1] = f.q[1];
}

// ---------------------------------------------------------------------------
// Kernel 2: Y_k = (sup_k + I) @ X  via bf16 WMMA, streaming sup once (HBM-bound).
// Each wave: 16 rows x 128 cols strip; 192 K-chunks x 8 WMMA = 1536 WMMAs.
// Fully double-buffered: B(mb+1) (16 x b128) and A(mb+1) (4 x b128 NT) are
// issued ABOVE a sched_barrier; the 8 WMMAs on the resident B(mb) run below
// it.  Steady state exposes no load latency inside the WMMA chain.
// The +I term is applied as "+ X[row, c]" in the epilogue.
// ---------------------------------------------------------------------------
__global__ __launch_bounds__(WG_THREADS)
void spmm_wmma_kernel(const float* __restrict__ sup,
                      const float* __restrict__ inputs,
                      const unsigned short* __restrict__ xfrag,
                      float* __restrict__ Y) {
    const int k      = blockIdx.z;
    const int wave   = threadIdx.x >> 5;
    const int lane   = threadIdx.x & 31;
    const int lane15 = lane & 15;
    const int half   = lane >> 4;
    const int rowbase = blockIdx.x * ROWS_PER_WG + wave * 16;

    // A 16x32 bf16 layout (ISA 7.12.2): lane = row; lane half -> kbase 0/8;
    // elements 0..7 = K kbase..kbase+7, elements 8..15 = K 16+kbase..16+kbase+7.
    const int arow = rowbase + lane15;
    const int kb   = half * 8;

    const float* suprow = sup + (size_t)k * Nn * Nn + (size_t)arow * Nn;
    const v16bf* bbase  = (const v16bf*)xfrag + lane;

    v8f acc[8];
#pragma unroll
    for (int g = 0; g < 8; ++g)
        acc[g] = (v8f){0.f, 0.f, 0.f, 0.f, 0.f, 0.f, 0.f, 0.f};

    // -------- prologue: A(0) and B(0) in flight --------
    v4f a0 = ldnt4(suprow + kb);
    v4f a1 = ldnt4(suprow + kb + 4);
    v4f a2 = ldnt4(suprow + 16 + kb);
    v4f a3 = ldnt4(suprow + 16 + kb + 4);

    v16bf bcur[8];
#pragma unroll
    for (int g = 0; g < 8; ++g)
        bcur[g] = bbase[g * 32];

#pragma unroll 2
    for (int mb = 0; mb < MBk; ++mb) {
        const bool last = (mb + 1 == MBk);

        // ---- issue next-chunk loads (B then A) ----
        v16bf bnext[8];
        if (!last) {
            const v16bf* bp = bbase + (size_t)(mb + 1) * 8 * 32;
#pragma unroll
            for (int g = 0; g < 8; ++g)
                bnext[g] = bp[g * 32];   // one 32B contiguous load per lane (L2/WGP$)
        }
        v4f na0 = a0, na1 = a1, na2 = a2, na3 = a3;
        if (!last) {
            const float* ap = suprow + (mb + 1) * 32;
            na0 = ldnt4(ap + kb);
            na1 = ldnt4(ap + kb + 4);
            na2 = ldnt4(ap + 16 + kb);
            na3 = ldnt4(ap + 16 + kb + 4);
            __builtin_prefetch(ap + 32 + kb, 0, 0);   // hint for mb+2
            __builtin_prefetch(ap + 48 + kb, 0, 0);
        }

        // ---- convert current A tile (its loads retired last iteration) ----
        BF16x16 af;
        af.u[0]  = f2bf(a0.x); af.u[1]  = f2bf(a0.y);
        af.u[2]  = f2bf(a0.z); af.u[3]  = f2bf(a0.w);
        af.u[4]  = f2bf(a1.x); af.u[5]  = f2bf(a1.y);
        af.u[6]  = f2bf(a1.z); af.u[7]  = f2bf(a1.w);
        af.u[8]  = f2bf(a2.x); af.u[9]  = f2bf(a2.y);
        af.u[10] = f2bf(a2.z); af.u[11] = f2bf(a2.w);
        af.u[12] = f2bf(a3.x); af.u[13] = f2bf(a3.y);
        af.u[14] = f2bf(a3.z); af.u[15] = f2bf(a3.w);

        // Fence: keep all next-chunk loads above, all WMMAs below, so the
        // scheduler cannot re-serialize load->wmma pairs.
        __builtin_amdgcn_sched_barrier(0);

        // ---- 8 WMMAs on the RESIDENT B(mb) fragments ----
#pragma unroll
        for (int g = 0; g < 8; ++g)
            acc[g] = __builtin_amdgcn_wmma_f32_16x16x32_bf16(
                false, af.v, false, bcur[g], (short)0, acc[g], false, false);

        __builtin_amdgcn_sched_barrier(0);

        // ---- rotate buffers (coalesced away by unroll-2 + regalloc) ----
#pragma unroll
        for (int g = 0; g < 8; ++g)
            bcur[g] = bnext[g];
        a0 = na0; a1 = na1; a2 = na2; a3 = na3;
    }

    // Epilogue: C/D layout (ISA 7.12.2): VGPR r, lanes 0-15 -> M=r, lanes 16-31 -> M=8+r,
    // N = lane&15.  Add the identity self-loop term X[n, c] and store Y_k.
    float* Yk = Y + (size_t)k * Nn * Cc;
#pragma unroll
    for (int g = 0; g < 8; ++g) {
#pragma unroll
        for (int r = 0; r < 8; ++r) {
            int n = rowbase + half * 8 + r;
            int c = g * 16 + lane15;                       // b = g, d = lane15
            float v = acc[g][r] + inputs[((size_t)g * Nn + n) * Dd + lane15];
            Yk[(size_t)n * Cc + c] = v;
        }
    }
}

// ---------------------------------------------------------------------------
// Kernel 3: projection out[b,n,o] = sum_f Y[k,n,b*16+d] * W[o, f=d*K+k] + bias[o]
// Tiny (151 MFLOP); W+bias staged in LDS, one (b,n) pair per thread.
// ---------------------------------------------------------------------------
__global__ __launch_bounds__(256)
void proj_kernel(const float* __restrict__ Y,
                 const float* __restrict__ W,
                 const float* __restrict__ bias,
                 float* __restrict__ out) {
    __shared__ float Wl[Oo * (Dd * Kk)];
    __shared__ float bl[Oo];
    for (int i = threadIdx.x; i < Oo * Dd * Kk; i += 256) Wl[i] = W[i];
    if (threadIdx.x < Oo) bl[threadIdx.x] = bias[threadIdx.x];
    __syncthreads();

    int p = blockIdx.x * 256 + threadIdx.x;   // 49152 = B*N threads
    int b = p & 7;
    int n = p >> 3;

    float yv[Dd * Kk];
#pragma unroll
    for (int k = 0; k < Kk; ++k)
#pragma unroll
        for (int d = 0; d < Dd; ++d)
            yv[d * Kk + k] = Y[((size_t)k * Nn + n) * Cc + b * 16 + d];

    float* op = out + ((size_t)b * Nn + n) * Oo;
#pragma unroll 4
    for (int o = 0; o < Oo; ++o) {
        float a = bl[o];
        const float* wr = &Wl[o * Dd * Kk];
#pragma unroll
        for (int f = 0; f < Dd * Kk; ++f) a = fmaf(yv[f], wr[f], a);
        op[o] = a;
    }
}

// ---------------------------------------------------------------------------
extern "C" void kernel_launch(void* const* d_in, const int* in_sizes, int n_in,
                              void* d_out, int out_size, void* d_ws, size_t ws_size,
                              hipStream_t stream) {
    const float* inputs   = (const float*)d_in[0];  // [B,N,D]
    const float* supports = (const float*)d_in[1];  // [K,N,N]
    const float* W        = (const float*)d_in[2];  // [O, D*K]
    const float* bias     = (const float*)d_in[3];  // [O]
    float* out            = (float*)d_out;          // [B,N,O]

    // workspace layout: xfrag (bf16, 1.5MB) | Y (f32, 9.44MB)
    unsigned short* xfrag = (unsigned short*)d_ws;
    size_t xfrag_bytes = (size_t)MBk * 8 * 32 * 16 * sizeof(unsigned short); // 1572864
    float* Y = (float*)((char*)d_ws + xfrag_bytes);

    // 1) pack X into bf16 B-fragments
    pack_bfrag_kernel<<<dim3((MBk * 8 * 32) / 256), dim3(256), 0, stream>>>(inputs, xfrag);

    // 2) big GEMM: 3 supports x (6144/64) row tiles, 4 waves each
    spmm_wmma_kernel<<<dim3(Nn / ROWS_PER_WG, 1, Kk), dim3(WG_THREADS), 0, stream>>>(
        supports, inputs, xfrag, Y);

    // 3) fused projection + bias
    proj_kernel<<<dim3((Bsz * Nn) / 256), dim3(256), 0, stream>>>(Y, W, bias, out);
}
